// FPConv6x6_BaseBlock_86517821212884
// MI455X (gfx1250) — matrix operand order repaired and checked by reading, not compiled, not run on hardware
//
#include <hip/hip_runtime.h>
#include <hip/hip_bf16.h>

typedef __attribute__((ext_vector_type(16))) _Float16 v16h;
typedef __attribute__((ext_vector_type(8)))  _Float16 v8h;
typedef __attribute__((ext_vector_type(8)))  float    v8f;

#define NEG     0.2f
#define RADIUS  0.1f
#define NSAMPLE 32
#define BQ   4
#define NP   8192
#define SQ   2048
#define CF   64
#define OUTC 128
#define CHUNK 2048

__device__ __forceinline__ float leaky(float x){ return x > 0.f ? x : NEG * x; }

// ---- WMMA fragment index maps (CDNA5 ISA 7.12.2, wave32) ----
// 16-bit A 16x32: row M = lane&15; K runs: {base, base+1..+7} with base = k0 + 8*(lane>=16) (+16 for h>=8)
__device__ __forceinline__ int b_kmap(int h, int lane){
  return ((lane >> 4) << 3) + (h & 7) + ((h >> 3) << 4);   // B 32x16: col N = lane&15; K = 8*(lane>=16) + (h&7) + 16*(h>=8)
}
// f32 D 16x16: vgpr r -> M = r + (lane>=16 ? 8 : 0), N = lane&15

// ================= kernel: copy new_xyz to head of d_out =================
__global__ void k_copy_newxyz(const float* __restrict__ nx, float* __restrict__ out, int n){
  int i = blockIdx.x * blockDim.x + threadIdx.x;
  if(i < n) out[i] = nx[i];
}

// ===== kernel: pack conv weights into WMMA B-fragment order, f16 =====
// K-ordering for all convs: k = t*64 + ic   (t = spatial tap, ic = input channel)
// Wf[( (kt*NT + nt)*32 + lane )*16 + h] = W[o = nt*16 + (lane&15)][k = kt*32 + b_kmap(h,lane)]
__global__ void k_pack_weights(const float* __restrict__ c1W, const float* __restrict__ c2W,
                               const float* __restrict__ c3W,
                               _Float16* __restrict__ Wf1, _Float16* __restrict__ Wf2,
                               _Float16* __restrict__ Wf3){
  int i = blockIdx.x * blockDim.x + threadIdx.x;
  if(i < 18 * 4 * 32 * 16){                    // conv1/conv2: K=576 (18 kt), N=64 (4 nt)
    int h = i & 15, lane = (i >> 4) & 31, nt = (i >> 9) & 3, kt = i >> 11;
    int o  = nt * 16 + (lane & 15);
    int k  = kt * 32 + b_kmap(h, lane);
    int ic = k & 63, t = k >> 6;               // t in 0..8
    int ky = t / 3, kx = t - ky * 3;
    Wf1[i] = (_Float16)c1W[(o * 64 + ic) * 9 + ky * 3 + kx];
    Wf2[i] = (_Float16)c2W[(o * 64 + ic) * 9 + ky * 3 + kx];
  }
  if(i < 8 * 8 * 32 * 16){                     // conv3: K=256 (8 kt), N=128 (8 nt)
    int h = i & 15, lane = (i >> 4) & 31, nt = (i >> 9) & 7, kt = i >> 12;
    int o  = nt * 16 + (lane & 15);
    int k  = kt * 32 + b_kmap(h, lane);
    int ic = k & 63, t = k >> 6;               // t = ky*2+kx in 0..3
    Wf3[i] = (_Float16)c3W[(o * 64 + ic) * 4 + t];
  }
}

// ================= kernel: ball query (LDS-tiled, wave32 ballot) =================
__global__ __launch_bounds__(256) void k_ballquery(
    const float* __restrict__ xyz,    // (B,N,3)
    const float* __restrict__ nxyz,   // (B,S,3)
    int*   __restrict__ idx_ws,       // (B,S,32)
    float* __restrict__ g_ws)         // (B,S,32,3)
{
  __shared__ float sx[CHUNK], sy[CHUNK], sz[CHUNK];
  __shared__ int   sidx[8][NSAMPLE];
  __shared__ float sgx[8][NSAMPLE], sgy[8][NSAMPLE], sgz[8][NSAMPLE];

  const int b    = blockIdx.x / (SQ / 8);
  const int s0   = (blockIdx.x % (SQ / 8)) * 8;
  const int w    = threadIdx.x >> 5;
  const int lane = threadIdx.x & 31;
  const int s    = s0 + w;

  const float qx = nxyz[((size_t)b * SQ + s) * 3 + 0];
  const float qy = nxyz[((size_t)b * SQ + s) * 3 + 1];
  const float qz = nxyz[((size_t)b * SQ + s) * 3 + 2];
  const float r2 = RADIUS * RADIUS;
  const float invR = 1.f / RADIUS;

  int cnt = 0;
  for(int c = 0; c < NP / CHUNK; ++c){
    const float* src = xyz + ((size_t)b * NP + (size_t)c * CHUNK) * 3;
    for(int t = threadIdx.x; t < CHUNK; t += 256){
      sx[t] = src[t * 3 + 0];
      sy[t] = src[t * 3 + 1];
      sz[t] = src[t * 3 + 2];
    }
    __syncthreads();
    if(cnt < NSAMPLE){
      for(int t = 0; t < CHUNK && cnt < NSAMPLE; t += 32){
        float dx = sx[t + lane] - qx;
        float dy = sy[t + lane] - qy;
        float dz = sz[t + lane] - qz;
        bool hit = (dx * dx + dy * dy + dz * dz) < r2;
        unsigned mask = (unsigned)__ballot(hit);
        int rank = __popc(mask & ((1u << lane) - 1u));
        int slot = cnt + rank;
        if(hit && slot < NSAMPLE){
          sidx[w][slot] = c * CHUNK + t + lane;
          sgx[w][slot]  = dx * invR;
          sgy[w][slot]  = dy * invR;
          sgz[w][slot]  = dz * invR;
        }
        cnt += __popc(mask);
      }
    }
    __syncthreads();
  }
  if(cnt > NSAMPLE) cnt = NSAMPLE;
  if(cnt == 0){                           // reference fallback: point 0
    if(lane == 0){
      float px = xyz[(size_t)b * NP * 3 + 0];
      float py = xyz[(size_t)b * NP * 3 + 1];
      float pz = xyz[(size_t)b * NP * 3 + 2];
      sidx[w][0] = 0;
      sgx[w][0] = (px - qx) * invR;
      sgy[w][0] = (py - qy) * invR;
      sgz[w][0] = (pz - qz) * invR;
    }
    cnt = 1;
  }
  __syncthreads();
  int   oi = (lane < cnt) ? sidx[w][lane] : sidx[w][0];
  float ox = (lane < cnt) ? sgx[w][lane]  : sgx[w][0];
  float oy = (lane < cnt) ? sgy[w][lane]  : sgy[w][0];
  float oz = (lane < cnt) ? sgz[w][lane]  : sgz[w][0];
  size_t base = ((size_t)b * SQ + s) * NSAMPLE + lane;
  idx_ws[base]       = oi;
  g_ws[base * 3 + 0] = ox;
  g_ws[base * 3 + 1] = oy;
  g_ws[base * 3 + 2] = oz;
}

// ============ fused kernel: MLP -> norm -> einsum -> conv stack (1 wave / (b,s)) ============
__global__ __launch_bounds__(32) void k_fused(
    const float* __restrict__ features,   // (B,C,N)
    const int*   __restrict__ idx_ws,
    const float* __restrict__ g_ws,
    const float* __restrict__ pnW0, const float* __restrict__ pnb0,
    const float* __restrict__ pnW1, const float* __restrict__ pnb1,
    const float* __restrict__ pnW2, const float* __restrict__ pnb2,
    const float* __restrict__ fuW0, const float* __restrict__ fub0,
    const float* __restrict__ fuW1, const float* __restrict__ fub1,
    const float* __restrict__ wtsW, const float* __restrict__ wtsb,
    const float* __restrict__ bias,
    const float* __restrict__ c1b, const float* __restrict__ c2b, const float* __restrict__ c3b,
    const _Float16* __restrict__ Wf1, const _Float16* __restrict__ Wf2,
    const _Float16* __restrict__ Wf3,
    float* __restrict__ outF)
{
  __shared__ float    biasL[64], c1bL[64], c2bL[64], c3bL[128];
  __shared__ float    wmat[36][32];
  __shared__ float    s2v[36];
  __shared__ _Float16 wh[48][32];        // padded einsum A (rows 36..47 = 0)
  __shared__ _Float16 fT[64][32];        // gathered features, [channel][sample]
  __shared__ _Float16 img0T[48][64];     // [spatial 6x6 (+pad)][channel]
  __shared__ _Float16 img1T[16][64];     // [spatial 4x4][channel]
  __shared__ _Float16 img2T[16][64];     // [spatial 2x2 (+pad)][channel]

  const int bs   = blockIdx.x;
  const int b    = bs / SQ, s = bs - b * SQ;
  const int lane = threadIdx.x;          // lane == sample index n
  const int col  = lane & 15;
  const int loff = (lane >> 4) << 3;     // K sub-run offset for A fragments
  const int kb16 = (lane >> 4) << 4;     // K offset for B fragments

  // ---------- preload biases ----------
  for(int i = lane; i < 64; i += 32){
    biasL[i] = bias[i];  c1bL[i] = c1b[i];  c2bL[i] = c2b[i];
  }
  for(int i = lane; i < 128; i += 32) c3bL[i] = c3b[i];

  // ---------- per-sample geometry ----------
  size_t gbase = (size_t)bs * NSAMPLE + lane;
  int   pid = idx_ws[gbase];
  float gx = g_ws[gbase * 3 + 0];
  float gy = g_ws[gbase * 3 + 1];
  float gz = g_ws[gbase * 3 + 2];

  // ---------- tiny MLP 3->8->16->16->16->32->36 (per lane) ----------
  float t8[8], h16a[16], h16b[16], h32[32];
  #pragma unroll
  for(int o = 0; o < 8; ++o){
    float acc = pnb0[o];
    acc += pnW0[o*3+0]*gx + pnW0[o*3+1]*gy + pnW0[o*3+2]*gz;
    t8[o] = fmaxf(acc, 0.f);
  }
  #pragma unroll
  for(int o = 0; o < 16; ++o){
    float acc = pnb1[o];
    #pragma unroll
    for(int i = 0; i < 8; ++i) acc += pnW1[o*8+i] * t8[i];
    h16a[o] = fmaxf(acc, 0.f);
  }
  #pragma unroll
  for(int o = 0; o < 16; ++o){
    float acc = pnb2[o];
    #pragma unroll
    for(int i = 0; i < 16; ++i) acc += pnW2[o*16+i] * h16a[i];
    h16b[o] = fmaxf(acc, 0.f);
  }
  #pragma unroll
  for(int o = 0; o < 16; ++o){
    float acc = fub0[o];
    #pragma unroll
    for(int i = 0; i < 16; ++i) acc += fuW0[o*16+i] * h16b[i];
    h16a[o] = fmaxf(acc, 0.f);
  }
  #pragma unroll
  for(int o = 0; o < 32; ++o){
    float acc = fub1[o];
    #pragma unroll
    for(int i = 0; i < 16; ++i) acc += fuW1[o*16+i] * h16a[i];
    h32[o] = fmaxf(acc, 0.f);
  }
  float ssq = 0.f;
  for(int o = 0; o < 36; ++o){
    float acc = wtsb[o];
    #pragma unroll
    for(int i = 0; i < 32; ++i) acc += wtsW[o*32+i] * h32[i];
    wmat[o][lane] = acc;                 // un-normalized w
    ssq += acc * acc;
  }
  float inv1 = 1.f / sqrtf(fmaxf(ssq, 1e-8f));   // per-sample (axis=1) norm
  __syncthreads();

  for(int m = lane; m < 36; m += 32){    // per-row (axis=3) norm from ORIGINAL w^2
    float rs = 0.f;
    for(int n = 0; n < 32; ++n){ float v = wmat[m][n]; rs += v * v; }
    s2v[m] = fmaxf(sqrtf(fmaxf(rs, 1e-8f)), 1.f);
  }
  __syncthreads();

  for(int m = 0; m < 36; ++m)
    wh[m][lane] = (_Float16)(wmat[m][lane] * inv1 / s2v[m]);
  for(int m = 36; m < 48; ++m)
    wh[m][lane] = (_Float16)0.f;

  // ---------- gather features: fT[c][n] ----------
  const float* fb = features + (size_t)b * CF * NP;
  for(int c = 0; c < CF; ++c)
    fT[c][lane] = (_Float16)fb[(size_t)c * NP + pid];
  __syncthreads();

  union frag { v16h v; v8h h[2]; };

  // ---------- einsum: w^T(48x32) @ f(32x64) -> pf + bias + leaky -> img0T ----------
  {
    v16h bfr[4];
    #pragma unroll
    for(int ct = 0; ct < 4; ++ct)
      bfr[ct] = *(const v16h*)&fT[ct*16 + col][kb16];   // contiguous 16 K-halfs
    for(int mt = 0; mt < 3; ++mt){
      frag af;
      af.v = *(const v16h*)&wh[mt*16 + col][kb16];
      #pragma unroll
      for(int ct = 0; ct < 4; ++ct){
        v8f acc = {};
        acc = __builtin_amdgcn_wmma_f32_16x16x32_f16(false, af.v, false, bfr[ct],
                                                     (short)0, acc, false, false);
        #pragma unroll
        for(int r = 0; r < 8; ++r){
          int m = mt*16 + r + ((lane >> 4) << 3);       // 0..47 (padded rows ok)
          int c = ct*16 + col;
          img0T[m][c] = (_Float16)leaky(acc[r] + biasL[c]);
        }
      }
    }
  }
  __syncthreads();

  // ---------- conv1: 6x6x64 -> 4x4x64 (M=16, K=576, N=64) ----------
  {
    const int py = col >> 2, px = col & 3;
    v8f acc[4] = {{}, {}, {}, {}};
    const v16h* Wf = (const v16h*)Wf1;
    for(int kt = 0; kt < 18; ++kt){
      frag af;
      #pragma unroll
      for(int half = 0; half < 2; ++half){
        int k  = kt*32 + half*16 + loff;
        int ic = k & 63, t = k >> 6;
        int ky = t / 3, kx = t - ky * 3;
        af.h[half] = *(const v8h*)&img0T[(py + ky)*6 + (px + kx)][ic];
      }
      #pragma unroll
      for(int nt = 0; nt < 4; ++nt)
        acc[nt] = __builtin_amdgcn_wmma_f32_16x16x32_f16(false, af.v, false,
                      Wf[(kt*4 + nt)*32 + lane], (short)0, acc[nt], false, false);
    }
    #pragma unroll
    for(int nt = 0; nt < 4; ++nt){
      int o = nt*16 + col;
      #pragma unroll
      for(int r = 0; r < 8; ++r){
        int p = r + ((lane >> 4) << 3);
        img1T[p][o] = (_Float16)leaky(acc[nt][r] + c1bL[o]);
      }
    }
  }
  __syncthreads();

  // ---------- conv2: 4x4x64 -> 2x2x64 (M=4 padded to 16, K=576, N=64) ----------
  {
    const int p4 = col & 3;                 // clamped position for safe addressing
    const int py = p4 >> 1, px = p4 & 1;
    const v16h zero = {};
    v8f acc[4] = {{}, {}, {}, {}};
    const v16h* Wf = (const v16h*)Wf2;
    for(int kt = 0; kt < 18; ++kt){
      frag af;
      #pragma unroll
      for(int half = 0; half < 2; ++half){
        int k  = kt*32 + half*16 + loff;
        int ic = k & 63, t = k >> 6;
        int ky = t / 3, kx = t - ky * 3;
        af.h[half] = *(const v8h*)&img1T[(py + ky)*4 + (px + kx)][ic];
      }
      af.v = (col < 4) ? af.v : zero;       // rows 4..15 of A are zero
      #pragma unroll
      for(int nt = 0; nt < 4; ++nt)
        acc[nt] = __builtin_amdgcn_wmma_f32_16x16x32_f16(false, af.v, false,
                      Wf[(kt*4 + nt)*32 + lane], (short)0, acc[nt], false, false);
    }
    #pragma unroll
    for(int nt = 0; nt < 4; ++nt){
      int o = nt*16 + col;
      #pragma unroll
      for(int r = 0; r < 8; ++r){
        int p = r + ((lane >> 4) << 3);     // 0..15, img2T padded
        img2T[p][o] = (_Float16)leaky(acc[nt][r] + c2bL[o]);
      }
    }
  }
  __syncthreads();

  // ---------- conv3: 2x2x64 -> 1x1x128 (M=1 padded to 16, K=256, N=128) ----------
  {
    const v16h zero = {};
    v8f acc[8] = {{}, {}, {}, {}, {}, {}, {}, {}};
    const v16h* Wf = (const v16h*)Wf3;
    for(int kt = 0; kt < 8; ++kt){
      frag af;
      #pragma unroll
      for(int half = 0; half < 2; ++half){
        int k  = kt*32 + half*16 + loff;
        int ic = k & 63, t = k >> 6;        // t in 0..3 == spatial index
        af.h[half] = *(const v8h*)&img2T[t][ic];
      }
      af.v = (col == 0) ? af.v : zero;      // only row M=0 is real
      #pragma unroll
      for(int nt = 0; nt < 8; ++nt)
        acc[nt] = __builtin_amdgcn_wmma_f32_16x16x32_f16(false, af.v, false,
                      Wf[(kt*8 + nt)*32 + lane], (short)0, acc[nt], false, false);
    }
    float* outP = outF + (size_t)BQ * SQ * 3;   // after new_xyz block
    if(lane < 16){                              // D row M=0 -> r=0, lanes 0..15
      #pragma unroll
      for(int nt = 0; nt < 8; ++nt){
        int o = nt*16 + lane;
        outP[((size_t)b * OUTC + o) * SQ + s] = leaky(acc[nt][0] + c3bL[o]);
      }
    }
  }
}

extern "C" void kernel_launch(void* const* d_in, const int* in_sizes, int n_in,
                              void* d_out, int out_size, void* d_ws, size_t ws_size,
                              hipStream_t stream) {
  const float* xyz   = (const float*)d_in[0];
  const float* nxyz  = (const float*)d_in[1];
  const float* feats = (const float*)d_in[2];
  const float* pnW0  = (const float*)d_in[3];
  const float* pnb0  = (const float*)d_in[4];
  const float* pnW1  = (const float*)d_in[5];
  const float* pnb1  = (const float*)d_in[6];
  const float* pnW2  = (const float*)d_in[7];
  const float* pnb2  = (const float*)d_in[8];
  const float* fuW0  = (const float*)d_in[9];
  const float* fub0  = (const float*)d_in[10];
  const float* fuW1  = (const float*)d_in[11];
  const float* fub1  = (const float*)d_in[12];
  const float* wtsW  = (const float*)d_in[13];
  const float* wtsb  = (const float*)d_in[14];
  const float* bias  = (const float*)d_in[15];
  const float* c1W   = (const float*)d_in[16];
  const float* c1b   = (const float*)d_in[17];
  const float* c2W   = (const float*)d_in[18];
  const float* c2b   = (const float*)d_in[19];
  const float* c3W   = (const float*)d_in[20];
  const float* c3b   = (const float*)d_in[21];
  float* out = (float*)d_out;

  // workspace layout (all offsets 32B-aligned)
  char* ws = (char*)d_ws;
  int*      idx_ws = (int*)ws;                                  // 1,048,576 B
  float*    g_ws   = (float*)(ws + (1 << 20));                  // 3,145,728 B
  _Float16* Wf1    = (_Float16*)(ws + 4194304);                 // 73,728 B
  _Float16* Wf2    = (_Float16*)(ws + 4268032);                 // 73,728 B
  _Float16* Wf3    = (_Float16*)(ws + 4341760);                 // 65,536 B

  int nCopy = BQ * SQ * 3;
  k_copy_newxyz<<<(nCopy + 255) / 256, 256, 0, stream>>>(nxyz, out, nCopy);

  k_pack_weights<<<(18 * 4 * 32 * 16 + 255) / 256, 256, 0, stream>>>(
      c1W, c2W, c3W, Wf1, Wf2, Wf3);

  k_ballquery<<<BQ * (SQ / 8), 256, 0, stream>>>(xyz, nxyz, idx_ws, g_ws);

  k_fused<<<BQ * SQ, 32, 0, stream>>>(
      feats, idx_ws, g_ws,
      pnW0, pnb0, pnW1, pnb1, pnW2, pnb2,
      fuW0, fub0, fuW1, fub1, wtsW, wtsb, bias,
      c1b, c2b, c3b, Wf1, Wf2, Wf3, out);
}
